// DevignModel_43482248905420
// MI455X (gfx1250) — matrix operand order repaired and checked by reading, not compiled, not run on hardware
//
#include <hip/hip_runtime.h>

#define NN    100000
#define NE    3200000
#define IND   128
#define HIDD  256
#define STEPS 6
#define NG    128
#define CONC  384

typedef __attribute__((ext_vector_type(16))) __bf16          v16bf;
typedef __attribute__((ext_vector_type(8)))  float           v8f;
typedef __attribute__((ext_vector_type(16))) unsigned short  v16us;
typedef __attribute__((ext_vector_type(8)))  unsigned short  v8us;
typedef __attribute__((ext_vector_type(4)))  unsigned short  v4us;

// ---------- helpers ----------
__device__ __forceinline__ unsigned short f2bf(float f) {
    unsigned int u = __builtin_bit_cast(unsigned int, f);
    u += 0x7FFFu + ((u >> 16) & 1u);           // round-to-nearest-even
    return (unsigned short)(u >> 16);
}

__device__ __forceinline__ v16bf mk_a(v8us lo, v8us hi) {
    v16us us = __builtin_shufflevector(lo, hi, 0, 1, 2, 3, 4, 5, 6, 7,
                                       8, 9, 10, 11, 12, 13, 14, 15);
    return __builtin_bit_cast(v16bf, us);
}

// ---------- weight packing into WMMA B-fragment order ----------
// Packed layout per matrix: tile index (nt*Ktiles + kt), then 32 lanes * 16 bf16.
// Lane L supplies B[k][n] with n = nt*16 + (L&15), k = kt*32 + (L>>4)*16 + j, j=0..15.

// ggnn_weight: 6 x [256,256] row-major, B[k][n] = W[k*256+n]
__global__ void pack_ggnn_kernel(const float* __restrict__ W, unsigned short* __restrict__ P) {
    int tid = blockIdx.x * 256 + threadIdx.x;          // 6*65536 total
    int j    =  tid        & 15;
    int lane = (tid >> 4)  & 31;
    int kt   = (tid >> 9)  & 7;
    int nt   = (tid >> 12) & 15;
    int t    =  tid >> 16;
    int k = kt * 32 + (lane >> 4) * 16 + j;
    int n = nt * 16 + (lane & 15);
    P[tid] = f2bf(W[t * 65536 + k * 256 + n]);
}

// w [Ntot,256] row-major, used as B = w^T : B[k][n] = w[n*256+k].  Ntot = 768.
__global__ void pack_wT_kernel(const float* __restrict__ W, unsigned short* __restrict__ P) {
    int tid = blockIdx.x * 256 + threadIdx.x;          // 8*48*512 = 196608 total
    int j    =  tid       & 15;
    int lane = (tid >> 4) & 31;
    int kt   = (tid >> 9) & 7;
    int nt   =  tid >> 12;                              // 0..47
    int k = kt * 32 + (lane >> 4) * 16 + j;
    int n = nt * 16 + (lane & 15);
    P[tid] = f2bf(W[n * 256 + k]);
}

// ---------- h = pad(x) to [NN, 256], f32 + bf16 copies ----------
__global__ void pad_h_kernel(const float* __restrict__ x, float* __restrict__ h,
                             unsigned short* __restrict__ hbf) {
    int tid = blockIdx.x * 256 + threadIdx.x;          // NN*256
    int i = tid >> 8, j = tid & 255;
    float v = (j < IND) ? x[i * IND + j] : 0.0f;
    h[tid] = v;
    hbf[tid] = f2bf(v);
}

__global__ void zero_f32_kernel(float* __restrict__ p, int n) {
    int tid = blockIdx.x * 256 + threadIdx.x;
    if (tid < n) p[tid] = 0.0f;
}

// f32 -> bf16, 4 elements per thread
__global__ void cvt_bf16_kernel(const float* __restrict__ in, unsigned short* __restrict__ out,
                                int n4) {
    int tid = blockIdx.x * 256 + threadIdx.x;
    if (tid >= n4) return;
    float4 f = ((const float4*)in)[tid];
    v4us o;
    o[0] = f2bf(f.x); o[1] = f2bf(f.y); o[2] = f2bf(f.z); o[3] = f2bf(f.w);
    ((v4us*)out)[tid] = o;
}

// ---------- WMMA GEMM: C[M,Ntot] = A_bf16[M,256] * Bpacked_bf16[256,Ntot] ----------
// grid.x = M/32 (row-pair tiles), grid.y = Ntiles/16, 256 threads = 8 waves.
// Each wave computes a 2x2 block of 16x16 output tiles: 2-way reuse of every
// A and B fragment -> 4 v_wmma per 8 b128 loads.
__global__ __launch_bounds__(256)
void gemm_wmma_kernel(const unsigned short* __restrict__ A, const unsigned short* __restrict__ Bp,
                      float* __restrict__ C, int Ntot) {
    const int wave = threadIdx.x >> 5;
    const int lane = threadIdx.x & 31;
    const int mt0 = blockIdx.x * 2;                     // two row tiles: mt0, mt0+1
    const int nt0 = (blockIdx.y * 8 + wave) * 2;        // two col tiles: nt0, nt0+1
    const int hh = lane >> 4;                           // half-select
    const int ln = lane & 15;

    const unsigned short* arow0 = A + (size_t)(mt0 * 16 + ln) * 256;
    const unsigned short* arow1 = arow0 + 16 * 256;
    const unsigned short* bp0   = Bp + ((size_t)nt0 * 8) * 512 + lane * 16;
    const unsigned short* bp1   = bp0 + 8 * 512;

    v8f acc00 = {}, acc01 = {}, acc10 = {}, acc11 = {};

#pragma unroll
    for (int kt = 0; kt < 8; ++kt) {
        // A fragments: elems 0..15 -> k = kt*32 + hh*16 + (0..15)
        v8us lo0 = *(const v8us*)(arow0 + kt * 32 + hh * 16);
        v8us hi0 = *(const v8us*)(arow0 + kt * 32 + 8 + hh * 16);
        v8us lo1 = *(const v8us*)(arow1 + kt * 32 + hh * 16);
        v8us hi1 = *(const v8us*)(arow1 + kt * 32 + 8 + hh * 16);
        v16bf a0 = mk_a(lo0, hi0);
        v16bf a1 = mk_a(lo1, hi1);

        v16bf b0 = *(const v16bf*)(bp0 + (size_t)kt * 512);
        v16bf b1 = *(const v16bf*)(bp1 + (size_t)kt * 512);

        acc00 = __builtin_amdgcn_wmma_f32_16x16x32_bf16(
            false, a0, false, b0, (short)0, acc00, false, false);
        acc01 = __builtin_amdgcn_wmma_f32_16x16x32_bf16(
            false, a0, false, b1, (short)0, acc01, false, false);
        acc10 = __builtin_amdgcn_wmma_f32_16x16x32_bf16(
            false, a1, false, b0, (short)0, acc10, false, false);
        acc11 = __builtin_amdgcn_wmma_f32_16x16x32_bf16(
            false, a1, false, b1, (short)0, acc11, false, false);
    }

    float* crow0 = C + (size_t)(mt0 * 16 + hh * 8) * Ntot + nt0 * 16 + ln;
    float* crow1 = crow0 + (size_t)16 * Ntot;
#pragma unroll
    for (int r = 0; r < 8; ++r) {
        crow0[(size_t)r * Ntot]      = acc00[r];
        crow0[(size_t)r * Ntot + 16] = acc01[r];
        crow1[(size_t)r * Ntot]      = acc10[r];
        crow1[(size_t)r * Ntot + 16] = acc11[r];
    }
}

// ---------- edge scatter-add: agg[dst] += m[src], wave-per-edge ----------
__global__ void scatter_add_kernel(const int* __restrict__ src, const int* __restrict__ dst,
                                   const float* __restrict__ m, float* __restrict__ agg, int E) {
    int gid = blockIdx.x * 256 + threadIdx.x;
    int e = gid >> 5;
    if (e >= E) return;
    int lane = gid & 31;
    int s = src[e], d = dst[e];
    const float4* mp = (const float4*)(m + (size_t)s * 256 + lane * 8);
    float4 a = mp[0], b = mp[1];
    float* ap = agg + (size_t)d * 256 + lane * 8;
    atomicAdd(ap + 0, a.x); atomicAdd(ap + 1, a.y);
    atomicAdd(ap + 2, a.z); atomicAdd(ap + 3, a.w);
    atomicAdd(ap + 4, b.x); atomicAdd(ap + 5, b.y);
    atomicAdd(ap + 6, b.z); atomicAdd(ap + 7, b.w);
}

// ---------- GRU cell elementwise (biases fused here) ----------
__global__ void gru_kernel(const float* __restrict__ gi, const float* __restrict__ gh,
                           const float* __restrict__ b_ih, const float* __restrict__ b_hh,
                           float* __restrict__ h, unsigned short* __restrict__ hbf) {
    int tid = blockIdx.x * 256 + threadIdx.x;          // NN*256
    int i = tid >> 8, c = tid & 255;
    size_t base = (size_t)i * 768;
    float ir = gi[base + c]       + b_ih[c];
    float iz = gi[base + 256 + c] + b_ih[256 + c];
    float in = gi[base + 512 + c] + b_ih[512 + c];
    float hr = gh[base + c]       + b_hh[c];
    float hz = gh[base + 256 + c] + b_hh[256 + c];
    float hn = gh[base + 512 + c] + b_hh[512 + c];
    float r = 1.0f / (1.0f + __expf(-(ir + hr)));
    float z = 1.0f / (1.0f + __expf(-(iz + hz)));
    float n = tanhf(in + r * hn);
    float ho = h[tid];
    float hv = (1.0f - z) * n + z * ho;
    h[tid] = hv;
    hbf[tid] = f2bf(hv);
}

// ---------- pooling ----------
__global__ void pool_feat_kernel(const float* __restrict__ x, const float* __restrict__ h,
                                 const int* __restrict__ batch, float* __restrict__ sums, int total) {
    int tid = blockIdx.x * 256 + threadIdx.x;          // NN*384
    if (tid >= total) return;
    int i = tid / CONC;
    int f = tid - i * CONC;
    float v = (f < IND) ? x[(size_t)i * IND + f] : h[(size_t)i * 256 + (f - IND)];
    atomicAdd(&sums[batch[i] * CONC + f], v);
}

__global__ void pool_cnt_kernel(const int* __restrict__ batch, float* __restrict__ cnt, int n) {
    int tid = blockIdx.x * 256 + threadIdx.x;
    if (tid < n) atomicAdd(&cnt[batch[tid]], 1.0f);
}

// ---------- MLP head: one block per graph ----------
__global__ void mlp_kernel(const float* __restrict__ sums, const float* __restrict__ cnt,
                           const float* __restrict__ w1, const float* __restrict__ b1,
                           const float* __restrict__ w2, const float* __restrict__ b2,
                           float* __restrict__ out) {
    __shared__ float sp[CONC];
    __shared__ float sh[CONC / 2];
    int g = blockIdx.x;
    float inv = 1.0f / fmaxf(cnt[g], 1.0f);
    for (int f = threadIdx.x; f < CONC; f += blockDim.x) sp[f] = sums[g * CONC + f] * inv;
    __syncthreads();
    int j = threadIdx.x;                                // 192 threads
    float acc = b1[j];
    for (int k = 0; k < CONC; ++k) acc += sp[k] * w1[j * CONC + k];
    sh[j] = fmaxf(acc, 0.0f);
    __syncthreads();
    if (j < 2) {
        float a = b2[j];
        for (int k = 0; k < CONC / 2; ++k) a += sh[k] * w2[j * (CONC / 2) + k];
        out[g * 2 + j] = a;
    }
}

// ---------- host ----------
static inline char* carve(char*& cur, size_t bytes) {
    char* p = cur;
    cur += (bytes + 255) & ~(size_t)255;
    return p;
}

extern "C" void kernel_launch(void* const* d_in, const int* in_sizes, int n_in,
                              void* d_out, int out_size, void* d_ws, size_t ws_size,
                              hipStream_t stream) {
    const float* x      = (const float*)d_in[0];
    const int*   ei     = (const int*)  d_in[1];
    const int*   batch  = (const int*)  d_in[2];
    const float* ggnn   = (const float*)d_in[3];
    const float* w_ih   = (const float*)d_in[4];
    const float* w_hh   = (const float*)d_in[5];
    const float* b_ih   = (const float*)d_in[6];
    const float* b_hh   = (const float*)d_in[7];
    const float* mlp_w1 = (const float*)d_in[8];
    const float* mlp_b1 = (const float*)d_in[9];
    const float* mlp_w2 = (const float*)d_in[10];
    const float* mlp_b2 = (const float*)d_in[11];
    float* out = (float*)d_out;

    const int* src = ei;
    const int* dst = ei + NE;

    char* cur = (char*)d_ws;
    float*          h     = (float*)carve(cur, (size_t)NN * 256 * 4);
    unsigned short* hbf   = (unsigned short*)carve(cur, (size_t)NN * 256 * 2);
    float*          agg   = (float*)carve(cur, (size_t)NN * 256 * 4);
    unsigned short* aggbf = (unsigned short*)carve(cur, (size_t)NN * 256 * 2);
    float*          bufA  = (float*)carve(cur, (size_t)NN * 768 * 4);  // m, then gi
    float*          bufB  = (float*)carve(cur, (size_t)NN * 768 * 4);  // gh
    unsigned short* p_ggnn = (unsigned short*)carve(cur, (size_t)6 * 65536 * 2);
    unsigned short* p_wih  = (unsigned short*)carve(cur, (size_t)196608 * 2);
    unsigned short* p_whh  = (unsigned short*)carve(cur, (size_t)196608 * 2);
    float* psum = (float*)carve(cur, (size_t)NG * CONC * 4);
    float* pcnt = (float*)carve(cur, (size_t)NG * 4);

    // pack weights to WMMA fragment order (bf16)
    pack_ggnn_kernel<<<(6 * 65536) / 256, 256, 0, stream>>>(ggnn, p_ggnn);
    pack_wT_kernel  <<<196608 / 256, 256, 0, stream>>>(w_ih, p_wih);
    pack_wT_kernel  <<<196608 / 256, 256, 0, stream>>>(w_hh, p_whh);

    // h = zero-pad(x), in f32 and bf16
    pad_h_kernel<<<NN, 256, 0, stream>>>(x, h, hbf);

    const int MT2 = NN / 32;                // 3125 row-pair tiles
    for (int t = 0; t < STEPS; ++t) {
        // m = h @ W_t   [NN,256]   (Ntiles=16 -> grid.y = 1)
        gemm_wmma_kernel<<<dim3(MT2, 1), 256, 0, stream>>>(hbf, p_ggnn + (size_t)t * 65536, bufA, 256);
        // agg = scatter_add(m[src] -> dst)
        zero_f32_kernel<<<(NN * 256) / 256, 256, 0, stream>>>(agg, NN * 256);
        scatter_add_kernel<<<(NE * 32) / 256, 256, 0, stream>>>(src, dst, bufA, agg, NE);
        // agg -> bf16 once
        cvt_bf16_kernel<<<(NN * 64) / 256, 256, 0, stream>>>(agg, aggbf, NN * 64);
        // gi = agg @ w_ih^T   [NN,768]  (Ntiles=48 -> grid.y = 3; overwrites dead m)
        gemm_wmma_kernel<<<dim3(MT2, 3), 256, 0, stream>>>(aggbf, p_wih, bufA, 768);
        // gh = h @ w_hh^T     [NN,768]
        gemm_wmma_kernel<<<dim3(MT2, 3), 256, 0, stream>>>(hbf, p_whh, bufB, 768);
        // h = GRU(agg-gates, h-gates, h), refresh bf16 copy
        gru_kernel<<<NN, 256, 0, stream>>>(bufA, bufB, b_ih, b_hh, h, hbf);
    }

    // global mean pool over graphs
    zero_f32_kernel<<<(NG * CONC + 255) / 256, 256, 0, stream>>>(psum, NG * CONC);
    zero_f32_kernel<<<1, 256, 0, stream>>>(pcnt, NG);
    pool_feat_kernel<<<((size_t)NN * CONC + 255) / 256, 256, 0, stream>>>(x, h, batch, psum, NN * CONC);
    pool_cnt_kernel<<<(NN + 255) / 256, 256, 0, stream>>>(batch, pcnt, NN);

    // MLP head -> logits [128,2]
    mlp_kernel<<<NG, CONC / 2, 0, stream>>>(psum, pcnt, mlp_w1, mlp_b1, mlp_w2, mlp_b2, out);
}
